// MemoryAttention_32993938768376
// MI455X (gfx1250) — matrix operand order repaired and checked by reading, not compiled, not run on hardware
//
#include <hip/hip_runtime.h>
#include <hip/hip_bf16.h>
#include <cstdint>

// ---------------------------------------------------------------------------
// Types for CDNA5 WMMA (wave32): v_wmma_f32_16x16x32_bf16
// ---------------------------------------------------------------------------
typedef __attribute__((ext_vector_type(16))) __bf16 v16bf;
typedef __attribute__((ext_vector_type(8)))  float  v8f;

union FragB16 {
    v16bf v;
    uint4 q[2];
};

// ---------------------------------------------------------------------------
// CDNA5 async copy: global -> LDS, 16B per lane, tracked by ASYNCcnt.
// lds_off = byte offset within the workgroup LDS allocation (flat addr[31:0]).
// ---------------------------------------------------------------------------
__device__ __forceinline__ void async_copy_b128(uint32_t lds_off, const void* gaddr) {
    asm volatile("global_load_async_to_lds_b128 %0, %1, off"
                 :: "v"(lds_off), "v"(gaddr)
                 : "memory");
}

template <int N>
__device__ __forceinline__ void wait_asynccnt() {
    asm volatile("s_wait_asynccnt %0" :: "i"(N) : "memory");
}

// Fragment fetch from LDS slabs (row stride = 32 bf16 = 64B).
// A operand (16x32): lane = row, 16 contiguous K values; kg = 16*(lane>>4)
__device__ __forceinline__ v16bf lds_frag_a(const __bf16* base, int row, int lane) {
    const int r  = lane & 15;
    const int kg = (lane >> 4) << 4;
    const __bf16* p = base + (row + r) * 32 + kg;
    FragB16 f;
    f.q[0] = *reinterpret_cast<const uint4*>(p);
    f.q[1] = *reinterpret_cast<const uint4*>(p + 8);
    return f.v;
}
// B operand (32x16): lane = column, 16 contiguous K values; kg = 16*(lane>>4)
__device__ __forceinline__ v16bf lds_frag_b(const __bf16* base, int col, int lane) {
    const int c  = lane & 15;
    const int kg = (lane >> 4) << 4;
    const __bf16* p = base + (col + c) * 32 + kg;
    FragB16 f;
    f.q[0] = *reinterpret_cast<const uint4*>(p);
    f.q[1] = *reinterpret_cast<const uint4*>(p + 8);
    return f.v;
}

// ---------------------------------------------------------------------------
// Elementwise f32 -> bf16 conversion
// ---------------------------------------------------------------------------
__global__ __launch_bounds__(256) void k_f32_to_bf16(const float* __restrict__ in,
                                                     __bf16* __restrict__ out, int n) {
    for (int i = blockIdx.x * blockDim.x + threadIdx.x; i < n;
         i += gridDim.x * blockDim.x) {
        out[i] = (__bf16)in[i];
    }
}

// ---------------------------------------------------------------------------
// Tiled transpose+convert: dst[c*rows + r] = (bf16) src[r*cols + c]
// rows, cols multiples of 32. blockIdx.z batches via element strides.
// ---------------------------------------------------------------------------
__global__ __launch_bounds__(256) void k_transpose_f32_bf16(
    const float* __restrict__ src, __bf16* __restrict__ dst,
    int rows, int cols, long sBatch, long dBatch) {
    __shared__ float tile[32][33];
    const float* s = src + (size_t)blockIdx.z * sBatch;
    __bf16*      d = dst + (size_t)blockIdx.z * dBatch;
    const int c0 = blockIdx.x * 32;
    const int r0 = blockIdx.y * 32;
    const int tx = threadIdx.x & 31;
    const int ty = threadIdx.x >> 5;   // 0..7
#pragma unroll
    for (int i = 0; i < 32; i += 8)
        tile[ty + i][tx] = s[(size_t)(r0 + ty + i) * cols + c0 + tx];
    __syncthreads();
#pragma unroll
    for (int i = 0; i < 32; i += 8)
        d[(size_t)(c0 + ty + i) * rows + r0 + tx] = (__bf16)tile[tx][ty + i];
}

// ---------------------------------------------------------------------------
// WMMA GEMM with async double-buffered LDS staging.
// C[Mr,N] = scale * (A[Mr,K] @ Bt[N,K]^T) + bias[N]
// Block tile 64(M) x 128(N), 8 waves (4x2), wave tile 16x64, K step 32.
// Per K step each thread issues 3x global_load_async_to_lds_b128.
// ---------------------------------------------------------------------------
__global__ __launch_bounds__(256) void k_gemm_bf16(
    const __bf16* __restrict__ A, const __bf16* __restrict__ Bt,
    const float* __restrict__ bias, float* __restrict__ C,
    __bf16* __restrict__ Cbf, int Mr, int N, int K,
    long batchA, long batchB, long batchC, float scale) {
    __shared__ __align__(16) __bf16 As[2][64][32];    // 4KB per buffer
    __shared__ __align__(16) __bf16 Bs[2][128][32];   // 8KB per buffer

    const int tid  = threadIdx.x;
    const int lane = tid & 31;
    const int wave = tid >> 5;
    const int wm   = (wave >> 1) << 4;   // 0,16,32,48
    const int wn   = (wave & 1) << 6;    // 0,64

    const int blockRow = blockIdx.x << 6;   // *64
    const int blockCol = blockIdx.y << 7;   // *128

    const __bf16* Ab = A  + (size_t)blockIdx.z * batchA;
    const __bf16* Bb = Bt + (size_t)blockIdx.z * batchB;

    // Per-thread async-copy assignments (16B segments).
    // A slab: 64 rows x 64B = 256 segs (1/thread); B slab: 128 rows x 64B = 512 segs (2/thread).
    const int aRow = tid >> 2, aPart = tid & 3;
    const int b0Seg = tid, b1Seg = tid + 256;
    const int b0Row = b0Seg >> 2, b0Part = b0Seg & 3;
    const int b1Row = b1Seg >> 2, b1Part = b1Seg & 3;

    const __bf16* gA  = Ab + (size_t)(blockRow + aRow) * K + aPart * 8;
    const __bf16* gB0 = Bb + (size_t)(blockCol + b0Row) * K + b0Part * 8;
    const __bf16* gB1 = Bb + (size_t)(blockCol + b1Row) * K + b1Part * 8;

    const uint32_t asBase = (uint32_t)(uintptr_t)(&As[0][0][0]);
    const uint32_t bsBase = (uint32_t)(uintptr_t)(&Bs[0][0][0]);
    const uint32_t lA  = asBase + (uint32_t)(aRow * 64 + aPart * 16);
    const uint32_t lB0 = bsBase + (uint32_t)(b0Row * 64 + b0Part * 16);
    const uint32_t lB1 = bsBase + (uint32_t)(b1Row * 64 + b1Part * 16);

    auto issue = [&](int kc, int buf) {
        const int koff = kc << 5;                        // *32 elements
        const uint32_t ab = (uint32_t)(buf * 4096);      // As buffer stride (bytes)
        const uint32_t bb = (uint32_t)(buf * 8192);      // Bs buffer stride (bytes)
        async_copy_b128(lA + ab,  gA  + koff);
        async_copy_b128(lB0 + bb, gB0 + koff);
        async_copy_b128(lB1 + bb, gB1 + koff);
    };

    v8f acc0 = {}, acc1 = {}, acc2 = {}, acc3 = {};

    const int nk = K >> 5;
    issue(0, 0);
    for (int kc = 0; kc < nk; ++kc) {
        const int cur = kc & 1;
        if (kc + 1 < nk) {
            issue(kc + 1, cur ^ 1);
            wait_asynccnt<3>();   // async loads complete in order: chunk kc has landed
        } else {
            wait_asynccnt<0>();
        }
        __syncthreads();          // all waves' slab pieces visible

        const __bf16* aB = &As[cur][0][0];
        const __bf16* bB = &Bs[cur][0][0];
        v16bf a  = lds_frag_a(aB, wm, lane);
        v16bf b0 = lds_frag_b(bB, wn,      lane);
        v16bf b1 = lds_frag_b(bB, wn + 16, lane);
        v16bf b2 = lds_frag_b(bB, wn + 32, lane);
        v16bf b3 = lds_frag_b(bB, wn + 48, lane);
        acc0 = __builtin_amdgcn_wmma_f32_16x16x32_bf16(false, a, false, b0, (short)0, acc0, false, false);
        acc1 = __builtin_amdgcn_wmma_f32_16x16x32_bf16(false, a, false, b1, (short)0, acc1, false, false);
        acc2 = __builtin_amdgcn_wmma_f32_16x16x32_bf16(false, a, false, b2, (short)0, acc2, false, false);
        acc3 = __builtin_amdgcn_wmma_f32_16x16x32_bf16(false, a, false, b3, (short)0, acc3, false, false);

        __syncthreads();          // protect buffer reuse before next issue
    }

    // C/D layout: VGPR r holds (M = r + 8*(lane>>4), N = lane&15)
    const int r0    = blockRow + wm + ((lane >> 4) << 3);
    const int cbase = blockCol + wn + (lane & 15);
    float*  Cb  = C   ? C   + (size_t)blockIdx.z * batchC : nullptr;
    __bf16* Cfb = Cbf ? Cbf + (size_t)blockIdx.z * batchC : nullptr;
#pragma unroll
    for (int r = 0; r < 8; ++r) {
        const size_t base = (size_t)(r0 + r) * N;
        float vals[4] = {acc0[r], acc1[r], acc2[r], acc3[r]};
#pragma unroll
        for (int j = 0; j < 4; ++j) {
            const int cc = cbase + j * 16;
            float v = vals[j] * scale + (bias ? bias[cc] : 0.0f);
            if (Cb)  Cb[base + cc]  = v;
            if (Cfb) Cfb[base + cc] = (__bf16)v;
        }
    }
}

// ---------------------------------------------------------------------------
// Relevance MLP fused into scores (in place):
// scores[b,s,m] *= (1/32) * sigmoid(b2 + sum_h relu(qp[b,s,h]+mp[b,m,h]+b1[h])*W2[h])
// Block: 128 threads (thread = m), 4 s-rows per block, q rows staged in LDS.
// ---------------------------------------------------------------------------
__global__ __launch_bounds__(128) void k_relevance(
    const float* __restrict__ qp, const float* __restrict__ mp,
    const float* __restrict__ b1, const float* __restrict__ W2,
    const float* __restrict__ b2, float* __restrict__ scores, int S) {
    constexpr int H  = 512;
    constexpr int Mm = 128;
    __shared__ float qs[4][H];
    const int b  = blockIdx.y;
    const int s0 = blockIdx.x * 4;
    const int m  = threadIdx.x;

    for (int idx = threadIdx.x; idx < 4 * H; idx += blockDim.x) {
        const int i = idx >> 9, h = idx & (H - 1);
        qs[i][h] = qp[(size_t)(b * S + s0 + i) * H + h];
    }
    __syncthreads();

    const float* mrow = mp + (size_t)(b * Mm + m) * H;
    float acc[4] = {0.f, 0.f, 0.f, 0.f};
    for (int h = 0; h < H; h += 4) {
        const float4 mv  = *reinterpret_cast<const float4*>(mrow + h);
        const float4 b1v = *reinterpret_cast<const float4*>(b1 + h);
        const float4 wv  = *reinterpret_cast<const float4*>(W2 + h);
        const float mb0 = mv.x + b1v.x, mb1 = mv.y + b1v.y;
        const float mb2 = mv.z + b1v.z, mb3 = mv.w + b1v.w;
#pragma unroll
        for (int i = 0; i < 4; ++i) {
            const float4 qv = *reinterpret_cast<const float4*>(&qs[i][h]);
            acc[i] += fmaxf(qv.x + mb0, 0.f) * wv.x
                    + fmaxf(qv.y + mb1, 0.f) * wv.y
                    + fmaxf(qv.z + mb2, 0.f) * wv.z
                    + fmaxf(qv.w + mb3, 0.f) * wv.w;
        }
    }
    const float bb = b2[0];
#pragma unroll
    for (int i = 0; i < 4; ++i) {
        const float rel = 1.0f / (1.0f + __expf(-(acc[i] + bb)));
        const size_t off = (size_t)(b * S + s0 + i) * Mm + m;
        scores[off] *= 0.03125f * rel;   // 1/sqrt(1024) fused here
    }
}

// ---------------------------------------------------------------------------
// Row softmax over M=128: one wave32 per row; writes f32 (final output) + bf16.
// ---------------------------------------------------------------------------
__global__ __launch_bounds__(256) void k_softmax(
    const float* __restrict__ sc, float* __restrict__ outw,
    __bf16* __restrict__ wbf, int rows) {
    constexpr int Mm = 128;
    const int lane = threadIdx.x & 31;
    const int wave = threadIdx.x >> 5;
    const int row  = blockIdx.x * 8 + wave;
    if (row >= rows) return;
    const float* r = sc + (size_t)row * Mm;
    float v[4];
    float mx = -3.402823e38f;
#pragma unroll
    for (int j = 0; j < 4; ++j) { v[j] = r[lane + 32 * j]; mx = fmaxf(mx, v[j]); }
#pragma unroll
    for (int off = 16; off > 0; off >>= 1) mx = fmaxf(mx, __shfl_xor(mx, off, 32));
    float sum = 0.f;
#pragma unroll
    for (int j = 0; j < 4; ++j) { v[j] = __expf(v[j] - mx); sum += v[j]; }
#pragma unroll
    for (int off = 16; off > 0; off >>= 1) sum += __shfl_xor(sum, off, 32);
    const float is = 1.0f / sum;
#pragma unroll
    for (int j = 0; j < 4; ++j) {
        const float w = v[j] * is;
        outw[(size_t)row * Mm + lane + 32 * j] = w;
        wbf [(size_t)row * Mm + lane + 32 * j] = (__bf16)w;
    }
}

// ---------------------------------------------------------------------------
// Host orchestration
// ---------------------------------------------------------------------------
extern "C" void kernel_launch(void* const* d_in, const int* in_sizes, int n_in,
                              void* d_out, int out_size, void* d_ws, size_t ws_size,
                              hipStream_t stream) {
    constexpr int B = 2, S = 1024, D = 1024, MD = 512, Mm = 128, H = 512;
    (void)in_sizes; (void)n_in; (void)out_size; (void)ws_size;

    const float* hs  = (const float*)d_in[0];
    const float* mem = (const float*)d_in[1];
    const float* Wq  = (const float*)d_in[2];
    const float* bq  = (const float*)d_in[3];
    const float* Wk  = (const float*)d_in[4];
    const float* bk  = (const float*)d_in[5];
    const float* Wv  = (const float*)d_in[6];
    const float* bv  = (const float*)d_in[7];
    const float* Wo  = (const float*)d_in[8];
    const float* bo  = (const float*)d_in[9];
    const float* W1  = (const float*)d_in[10];
    const float* b1  = (const float*)d_in[11];
    const float* W2  = (const float*)d_in[12];
    const float* b2  = (const float*)d_in[13];

    float* out_f32  = (float*)d_out;                       // (B,S,D)
    float* out_attn = (float*)d_out + (size_t)B * S * D;   // (B,S,Mm)

    // Workspace bump allocator (256B aligned)
    char*  w   = (char*)d_ws;
    size_t off = 0;
    auto alloc = [&](size_t bytes) -> char* {
        char* p = w + off;
        off = (off + bytes + 255) & ~(size_t)255;
        return p;
    };
    __bf16* hs_bf    = (__bf16*)alloc((size_t)B * S * D * 2);
    __bf16* mem_bf   = (__bf16*)alloc((size_t)B * Mm * MD * 2);
    __bf16* WqT      = (__bf16*)alloc((size_t)D * D * 2);    // N=D x K=D
    __bf16* WkT      = (__bf16*)alloc((size_t)D * MD * 2);   // N=D x K=MD
    __bf16* WvT      = (__bf16*)alloc((size_t)D * MD * 2);
    __bf16* WoT      = (__bf16*)alloc((size_t)D * D * 2);
    __bf16* W1qT     = (__bf16*)alloc((size_t)H * D * 2);    // N=H x K=D
    __bf16* W1mT     = (__bf16*)alloc((size_t)H * MD * 2);   // N=H x K=MD
    __bf16* Qbf      = (__bf16*)alloc((size_t)B * S * D * 2);
    __bf16* Kbf      = (__bf16*)alloc((size_t)B * Mm * D * 2);
    float*  Vf32     = (float*) alloc((size_t)B * Mm * D * 4);
    __bf16* VtBf     = (__bf16*)alloc((size_t)B * D * Mm * 2);
    float*  qpart    = (float*) alloc((size_t)B * S * H * 4);
    float*  mpart    = (float*) alloc((size_t)B * Mm * H * 4);
    float*  scores   = (float*) alloc((size_t)B * S * Mm * 4);
    __bf16* attnBf   = (__bf16*)alloc((size_t)B * S * Mm * 2);
    __bf16* aoutBf   = (__bf16*)alloc((size_t)B * S * D * 2);

    // 1) Activations -> bf16
    k_f32_to_bf16<<<1024, 256, 0, stream>>>(hs,  hs_bf,  B * S * D);
    k_f32_to_bf16<<<256,  256, 0, stream>>>(mem, mem_bf, B * Mm * MD);

    // 2) Weight transposes (K x N row-major -> N x K bf16)
    auto T = [&](const float* src, __bf16* dst, int rows, int cols) {
        k_transpose_f32_bf16<<<dim3(cols / 32, rows / 32, 1), 256, 0, stream>>>(
            src, dst, rows, cols, 0, 0);
    };
    T(Wq, WqT, D, D);
    T(Wk, WkT, MD, D);
    T(Wv, WvT, MD, D);
    T(Wo, WoT, D, D);
    T(W1,                 W1qT, D,  H);   // rows 0..1023 of W1
    T(W1 + (size_t)D * H, W1mT, MD, H);   // rows 1024..1535 of W1

    // GEMM launcher: block tile 64x128, grid (Mr/64, N/128, batch)
    auto gemm = [&](const __bf16* A, const __bf16* Bt, const float* bias,
                    float* C, __bf16* Cbf, int Mr, int N, int K,
                    long bA, long bB, long bC, float scale, int nz) {
        k_gemm_bf16<<<dim3(Mr / 64, N / 128, nz), 256, 0, stream>>>(
            A, Bt, bias, C, Cbf, Mr, N, K, bA, bB, bC, scale);
    };

    // 3) Projections
    gemm(hs_bf,  WqT, bq, nullptr, Qbf,  B * S,  D, D,  0, 0, 0, 1.0f, 1);  // queries
    gemm(mem_bf, WkT, bk, nullptr, Kbf,  B * Mm, D, MD, 0, 0, 0, 1.0f, 1);  // keys
    gemm(mem_bf, WvT, bv, Vf32,  nullptr, B * Mm, D, MD, 0, 0, 0, 1.0f, 1); // values (f32)

    // 4) Relevance-MLP inputs
    gemm(Qbf,    W1qT, nullptr, qpart, nullptr, B * S,  H, D,  0, 0, 0, 1.0f, 1);
    gemm(mem_bf, W1mT, nullptr, mpart, nullptr, B * Mm, H, MD, 0, 0, 0, 1.0f, 1);

    // 5) Raw scores: per batch Q(1024xD) @ K^T(Mm x D) -> (1024 x Mm)
    gemm(Qbf, Kbf, nullptr, scores, nullptr, S, Mm, D,
         (long)S * D, (long)Mm * D, (long)S * Mm, 1.0f, B);

    // 6) relevance * scale fused into scores (in place)
    k_relevance<<<dim3(S / 4, B, 1), 128, 0, stream>>>(qpart, mpart, b1, W2, b2,
                                                       scores, S);

    // 7) softmax -> attn weights (f32 into d_out, bf16 for next GEMM)
    k_softmax<<<(B * S) / 8, 256, 0, stream>>>(scores, out_attn, attnBf, B * S);

    // 8) V -> V^T bf16 per batch (D x Mm)
    k_transpose_f32_bf16<<<dim3(D / 32, Mm / 32, B), 256, 0, stream>>>(
        Vf32, VtBf, Mm, D, (long)Mm * D, (long)D * Mm);

    // 9) attn_output = attn @ V   (per batch: (S x Mm) @ (Mm x D))
    gemm(attnBf, VtBf, nullptr, nullptr, aoutBf, S, D, Mm,
         (long)S * Mm, (long)D * Mm, (long)S * D, 1.0f, B);

    // 10) out = attn_output @ Wo + bo  (written straight to d_out)
    gemm(aoutBf, WoT, bo, out_f32, nullptr, B * S, D, D, 0, 0, 0, 1.0f, 1);
}
